// PixelRNNLayer_23768349016230
// MI455X (gfx1250) — matrix operand order repaired and checked by reading, not compiled, not run on hardware
//
#include <hip/hip_runtime.h>
#include <hip/hip_bf16.h>
#include <math.h>

typedef __attribute__((ext_vector_type(16))) __bf16 v16bf;
typedef __attribute__((ext_vector_type(8)))  __bf16 v8bf;
typedef __attribute__((ext_vector_type(8)))  float  v8f;

// Problem constants
#define BATCH 8
#define CIN   96
#define OCH   96
#define GCH   384
#define HH    128
#define WW    128
#define NKT   9            // 288 / 32 k-steps per conv
#define NMT   24           // 384 / 16 m-tiles

#define PITCH 104          // bf16 per LDS row (208 B = 52 dwords)
#define NROWS 130          // x' = x + tap; halo rows 0 and 129 stay zero
#define BUFE  (NROWS * PITCH)

#define FRAG_ELEMS 512                      // 32 lanes * 16 bf16 per packed A fragment
#define PACK_ELEMS (NMT * NKT * FRAG_ELEMS)

// ---------------------------------------------------------------------------
// Prep: mask + bf16-convert + pack weights into WMMA A-fragment layout.
// A-fragment (16-bit, 16x32): lane l -> M = l%16 ; element j -> K = kmapA(j,l/16)
//   kmapA(j,h) = j + 16*h       (contiguous, matches B fragment)
// ---------------------------------------------------------------------------
__device__ __forceinline__ void decode_pack(int e, int& m, int& k) {
    int fragId = e >> 9;
    int r      = e & 511;
    int lane   = r >> 4;
    int j      = r & 15;
    int mTile  = fragId / NKT;
    int kTile  = fragId - NKT * mTile;
    int hh     = lane >> 4;
    m = mTile * 16 + (lane & 15);
    k = kTile * 32 + j + 16 * hh;
}

__global__ void prep_pack_kernel(const float* __restrict__ Wis,
                                 const float* __restrict__ bis,
                                 const float* __restrict__ Wss,
                                 const float* __restrict__ bss,
                                 __bf16* __restrict__ packIs,
                                 __bf16* __restrict__ packSs,
                                 float* __restrict__ bias) {
    int idx = blockIdx.x * blockDim.x + threadIdx.x;
    if (idx < PACK_ELEMS) {
        int m, k; decode_pack(idx, m, k);
        int tap = k / 96;
        int cin = k - 96 * tap;
        float w = Wis[(m * CIN + cin) * 3 + tap];   // W_is: (384, 96, 1, 3)
        if (tap == 2) {
            w = 0.f;                                 // right of center: zero
        } else if (tap == 1) {
            int colorOut = m / 128;                  // triangular RGB mask
            if (cin >= (colorOut + 1) * 32) w = 0.f;
        }
        packIs[idx] = (__bf16)w;
    } else if (idx < 2 * PACK_ELEMS) {
        int e = idx - PACK_ELEMS;
        int m, k; decode_pack(e, m, k);
        int tap = k / 96;
        int cin = k - 96 * tap;
        packSs[e] = (__bf16)Wss[(m * CIN + cin) * 3 + tap];  // W_ss: (384, 96, 3)
    } else if (idx < 2 * PACK_ELEMS + GCH) {
        int i = idx - 2 * PACK_ELEMS;
        bias[i] = bis[i] + bss[i];
    }
}

// ---------------------------------------------------------------------------
// Scan-kernel helpers
// ---------------------------------------------------------------------------
struct A4 { v16bf a0, a1, a2, a3; };   // A fragments for the 4 gates (g stride = 2 m-tiles = 18 frags)

__device__ __forceinline__ A4 loadA4(const __bf16* __restrict__ base, int laneOff, int kt) {
    A4 r;
    r.a0 = *(const v16bf*)(base + laneOff + ((kt +  0) << 9));
    r.a1 = *(const v16bf*)(base + laneOff + ((kt + 18) << 9));
    r.a2 = *(const v16bf*)(base + laneOff + ((kt + 36) << 9));
    r.a3 = *(const v16bf*)(base + laneOff + ((kt + 54) << 9));
    return r;
}

__device__ __forceinline__ v16bf loadBfrag(const __bf16* __restrict__ buf, int xb, int kt, int hh16) {
    int kbase = kt * 32 + hh16;          // 16-wide K group never crosses a tap
    int tap   = kbase / 96;
    int cin0  = kbase - 96 * tap;
    return *(const v16bf*)(buf + (xb + tap) * PITCH + cin0);
}

__device__ __forceinline__ void wmma4(v8f acc[4], const A4& a, v16bf b) {
    acc[0] = __builtin_amdgcn_wmma_f32_16x16x32_bf16(false, a.a0, false, b, (short)0, acc[0], false, false);
    acc[1] = __builtin_amdgcn_wmma_f32_16x16x32_bf16(false, a.a1, false, b, (short)0, acc[1], false, false);
    acc[2] = __builtin_amdgcn_wmma_f32_16x16x32_bf16(false, a.a2, false, b, (short)0, acc[2], false, false);
    acc[3] = __builtin_amdgcn_wmma_f32_16x16x32_bf16(false, a.a3, false, b, (short)0, acc[3], false, false);
}

// ---------------------------------------------------------------------------
// Fused scan: one workgroup per batch element, persistent over 128 rows.
// 16 waves = (half, xt). Per hc-tile: 4 gate accumulators; software-pipelined
// A (global, SGPR base + immediate offsets) and B (LDS, contiguous 32B) frags.
// ---------------------------------------------------------------------------
__global__ void __launch_bounds__(512, 1)
pixelrnn_scan_kernel(const float* __restrict__ X,
                     const __bf16* __restrict__ packIs,
                     const __bf16* __restrict__ packSs,
                     const float* __restrict__ bias,
                     float* __restrict__ out) {
    extern __shared__ char smem[];
    __bf16* sXT   = (__bf16*)smem;               // [2][BUFE]
    __bf16* sHT   = sXT + 2 * BUFE;              // [2][BUFE]
    float*  sbias = (float*)(sHT + 2 * BUFE);    // [GCH]

    const int b    = blockIdx.x;
    const int tid  = threadIdx.x;
    const int lane = tid & 31;
    const int wave = tid >> 5;
    const int hh   = lane >> 4;
    const int hh16 = hh * 16;
    const int ln   = lane & 15;
    const int xt   = wave & 7;
    const int half = wave >> 3;
    const int x0   = xt * 16;
    const int xb   = x0 + ln;        // B-fragment row base (before +tap)
    const int laneOff = lane * 16;   // per-lane A-fragment offset (elements)

    // one-time init: zero all buffers (h0 = 0 + halos), stage folded bias
    {
        unsigned int* z = (unsigned int*)smem;
        const int ndw = (4 * BUFE * 2) / 4;
        for (int t = tid; t < ndw; t += 512) z[t] = 0u;
        for (int t = tid; t < GCH; t += 512) sbias[t] = bias[t];
    }

    float cacc[3][8];
    #pragma unroll
    for (int t3 = 0; t3 < 3; ++t3)
        #pragma unroll
        for (int r = 0; r < 8; ++r) cacc[t3][r] = 0.f;

    const float* Xb = X + (size_t)b * CIN * HH * WW;
    __syncthreads();

    for (int y = 0; y < HH; ++y) {
        asm volatile("" ::: "memory");   // keep weight loads inside the row loop

        const int cur = y & 1;
        __bf16* sXcur = sXT + cur * BUFE;
        const __bf16* hIn  = sHT + cur * BUFE;
        __bf16*       hOut = sHT + (cur ^ 1) * BUFE;

        // ---- stage X row y transposed [x+1][cin] as bf16 ----------------
        for (int t = tid; t < CIN * (WW / 4); t += 512) {
            int cch = t >> 5;
            int x4  = t & 31;
            float4 v = *(const float4*)&Xb[(cch * HH + y) * WW + x4 * 4];
            __bf16* dst = sXcur + (x4 * 4 + 1) * PITCH + cch;
            dst[0 * PITCH] = (__bf16)v.x;
            dst[1 * PITCH] = (__bf16)v.y;
            dst[2 * PITCH] = (__bf16)v.z;
            dst[3 * PITCH] = (__bf16)v.w;
        }
        __syncthreads();   // the only barrier per row (double-buffered LDS)

        // ---- 3 hc-tiles, 4 gates each -----------------------------------
        #pragma unroll
        for (int t3 = 0; t3 < 3; ++t3) {
            const int hc0 = (half * 3 + t3) * 16;
            // wave-uniform scalars: scalar branches (EXEC stays all-ones) and
            // SGPR A-fragment bases with immediate per-(gate,kt) offsets
            const int colorU = __builtin_amdgcn_readfirstlane(hc0 >> 5);
            const int m0u    = __builtin_amdgcn_readfirstlane(colorU * 128 + (hc0 & 31));
            const __bf16* pi = packIs + (((m0u >> 4) * NKT) << 9);
            const __bf16* ps = packSs + (((m0u >> 4) * NKT) << 9);

            v8f acc[4];
            #pragma unroll
            for (int g = 0; g < 4; ++g)
                acc[g] = *(const v8f*)&sbias[m0u + g * 32 + 8 * hh];

            A4 ac, an; v16bf bc, bn;

            // ---- input-to-state (k-steps 0..3 always; +color extra) -----
            ac = loadA4(pi, laneOff, 0); bc = loadBfrag(sXcur, xb, 0, hh16);
            an = loadA4(pi, laneOff, 1); bn = loadBfrag(sXcur, xb, 1, hh16);
            wmma4(acc, ac, bc); ac = an; bc = bn;                       // kt 0
            an = loadA4(pi, laneOff, 2); bn = loadBfrag(sXcur, xb, 2, hh16);
            wmma4(acc, ac, bc); ac = an; bc = bn;                       // kt 1
            an = loadA4(pi, laneOff, 3); bn = loadBfrag(sXcur, xb, 3, hh16);
            wmma4(acc, ac, bc); ac = an; bc = bn;                       // kt 2
            if (colorU > 0) { an = loadA4(pi, laneOff, 4); bn = loadBfrag(sXcur, xb, 4, hh16); }
            wmma4(acc, ac, bc);                                         // kt 3
            if (colorU > 0) {
                ac = an; bc = bn;
                if (colorU > 1) { an = loadA4(pi, laneOff, 5); bn = loadBfrag(sXcur, xb, 5, hh16); }
                wmma4(acc, ac, bc);                                     // kt 4
                if (colorU > 1) wmma4(acc, an, bn);                     // kt 5
            }

            // ---- state-to-state: full K = 288, depth-1 pipeline ---------
            ac = loadA4(ps, laneOff, 0); bc = loadBfrag(hIn, xb, 0, hh16);
            #pragma unroll
            for (int kt = 0; kt < NKT; ++kt) {
                A4 anx; v16bf bnx;
                if (kt + 1 < NKT) { anx = loadA4(ps, laneOff, kt + 1); bnx = loadBfrag(hIn, xb, kt + 1, hh16); }
                wmma4(acc, ac, bc);
                if (kt + 1 < NKT) { ac = anx; bc = bnx; }
            }

            // ---- LSTM pointwise; h -> out (f32) + hOut (bf16 b128) ------
            v8bf hw;
            const int x = xb;
            #pragma unroll
            for (int r = 0; r < 8; ++r) {
                float fg = 1.f / (1.f + __expf(-acc[0][r]));
                float ig = 1.f / (1.f + __expf(-acc[1][r]));
                float og = 1.f / (1.f + __expf(-acc[2][r]));
                float gg = tanhf(acc[3][r]);
                float cn = fg * cacc[t3][r] + ig * gg;
                cacc[t3][r] = cn;
                float hv = og * tanhf(cn);
                int hc = hc0 + r + 8 * hh;       // D layout: M = r + 8*(lane>=16)
                out[(((size_t)b * OCH + hc) * HH + y) * WW + x] = hv;
                hw[r] = (__bf16)hv;
            }
            *(v8bf*)&hOut[(x + 1) * PITCH + hc0 + 8 * hh] = hw;
        }
        // next row's post-stage barrier orders hOut writes vs. next reads
    }
}

// ---------------------------------------------------------------------------
extern "C" void kernel_launch(void* const* d_in, const int* in_sizes, int n_in,
                              void* d_out, int out_size, void* d_ws, size_t ws_size,
                              hipStream_t stream) {
    const float* X   = (const float*)d_in[0];   // (8, 96, 128, 128)
    const float* Wis = (const float*)d_in[1];   // (384, 96, 1, 3)
    const float* bis = (const float*)d_in[2];   // (384,)
    const float* Wss = (const float*)d_in[3];   // (384, 96, 3)
    const float* bss = (const float*)d_in[4];   // (384,)
    float* out = (float*)d_out;                 // (8, 96, 128, 128)

    __bf16* packIs = (__bf16*)d_ws;
    __bf16* packSs = (__bf16*)((char*)d_ws + (size_t)PACK_ELEMS * 2);
    float*  bias   = (float*)((char*)d_ws + (size_t)2 * PACK_ELEMS * 2);

    int prepTotal = 2 * PACK_ELEMS + GCH;
    prep_pack_kernel<<<(prepTotal + 511) / 512, 512, 0, stream>>>(
        Wis, bis, Wss, bss, packIs, packSs, bias);

    size_t ldsBytes = (size_t)4 * BUFE * 2 + GCH * 4;   // ~110 KB
    pixelrnn_scan_kernel<<<BATCH, 512, ldsBytes, stream>>>(X, packIs, packSs, bias, out);
}
